// LSTM_37082747633789
// MI455X (gfx1250) — matrix operand order repaired and checked by reading, not compiled, not run on hardware
//
#include <hip/hip_runtime.h>
#include <hip/hip_bf16.h>
#include <math.h>

// ---------------------------------------------------------------------------
// LSTM on MI455X (gfx1250, wave32) -- split-bf16 WMMA path.
//
// Roofline: 137 GFLOP total. The f32 WMMA (16x16x4) path costs 128 WMMA/tile;
// bf16 (16x16x32) with a hi/lo split costs 48 WMMA/tile at ~fp32 accuracy
// (missing only the lo*lo term ~2^-17). In the latency-bound recurrence this
// turns 8 dependency chains of depth 64 into 12 chains of depth 16 per wave.
//
//  Phase 0a: split X (and hx) into bf16 hi/lo, row-major.
//  Phase 0b: pack wih/whh into bf16 WMMA B-frag records (hi/lo).
//  Phase 1 : x_proj = X @ wih^T + bias -> f32 frag layout (1 block = 1 t).
//  Phase 2 : persistent scan; wave owns a 16x16 h-tile + its 4 gate tiles;
//            h kept as bf16 hi/lo (split by the producer, in registers).
// ---------------------------------------------------------------------------

typedef __bf16 bf16_t;
typedef __bf16 v8bf  __attribute__((ext_vector_type(8)));
typedef __bf16 v16bf __attribute__((ext_vector_type(16)));
typedef float  v4f   __attribute__((ext_vector_type(4)));
typedef float  v8f   __attribute__((ext_vector_type(8)));

#define T_STEPS 512
#define BATCH   64
#define ISZ     512
#define HSZ     512
#define NCT     128                        // col tiles (2048/16)
#define NKT     16                         // K tiles of 32 (512/32)
#define WPACK_ELEMS (NCT * NKT * 32 * 16)  // 1,048,576 bf16 per hi/lo array

// bf16 WMMA 16x16x32: D(16x16 f32) = A(16x32 bf16) * B(32x16 bf16) + C
static __device__ __forceinline__ v8f wmma_bf16(v16bf a, v16bf b, v8f c) {
    return __builtin_amdgcn_wmma_f32_16x16x32_bf16(
        false, a, false, b, (short)0, c, false, false);
}

static __device__ __forceinline__ v16bf join16(v8bf a, v8bf b) {
    return __builtin_shufflevector(a, b, 0, 1, 2, 3, 4, 5, 6, 7,
                                         8, 9, 10, 11, 12, 13, 14, 15);
}

// ---------------------------------------------------------------------------
// Phase 0a: elementwise split f32 -> bf16 hi + bf16 lo (residual).
// ---------------------------------------------------------------------------
__global__ __launch_bounds__(256, 1)
void split_f32_bf16(const float* __restrict__ src, bf16_t* __restrict__ hi,
                    bf16_t* __restrict__ lo, int n) {
    const int i = blockIdx.x * 256 + threadIdx.x;
    if (i < n) {
        const float x = src[i];
        const bf16_t h = (bf16_t)x;
        hi[i] = h;
        lo[i] = (bf16_t)(x - (float)h);
    }
}

// ---------------------------------------------------------------------------
// Phase 0b: pack W[2048 x 512] into bf16 B-frag records (hi and lo).
// Record (ct, kt): 32 lanes x 16 bf16, flat = ((ct*NKT + kt)*32 + lane)*16 + e
//   k = kt*32 + 16*(lane>>4) + e ;  n = ct*16 + (lane&15) ; value = W[n*512+k]
// (Contiguous k order, same as A frags: lanes 0-15 hold K=0-15, lanes 16-31 hold K=16-31.)
// ---------------------------------------------------------------------------
__global__ __launch_bounds__(256, 1)
void pack_w_bf16(const float* __restrict__ W, bf16_t* __restrict__ hi,
                 bf16_t* __restrict__ lo) {
    const int idx  = blockIdx.x * 256 + threadIdx.x;   // [0, WPACK_ELEMS)
    const int e    = idx & 15;
    const int lane = (idx >> 4) & 31;
    const int kt   = (idx >> 9) & 15;
    const int ct   = idx >> 13;
    const int k = kt * 32 + (lane >> 4) * 16 + e;
    const int n = ct * 16 + (lane & 15);
    const float x = W[(long)n * 512 + k];
    const bf16_t h = (bf16_t)x;
    hi[idx] = h;
    lo[idx] = (bf16_t)(x - (float)h);
}

// A-frag loader: 16-bit A layout. Lane holds row m=laneM; element e in
// [0,16) -> K = kb + 16*laneHi + e (matches packed B records).
static __device__ __forceinline__ v16bf load_afrag(const bf16_t* row, int kb,
                                                   int laneHi) {
    const bf16_t* p = row + kb + 16 * laneHi;
    v8bf a0 = *(const v8bf*)(p);
    v8bf a1 = *(const v8bf*)(p + 8);
    return join16(a0, a1);
}

static __device__ __forceinline__ v16bf load_bfrag(const bf16_t* rec) {
    v8bf b0 = *(const v8bf*)(rec);
    v8bf b1 = *(const v8bf*)(rec + 8);
    return join16(b0, b1);
}

// ---------------------------------------------------------------------------
// Phase 1: X * wih^T -> Pf (f32 frag layout), + bias.
// grid = (512, 32); block = 128 (4 waves). blockIdx.x == timestep t.
// Wave = batch tile r; block covers 4 col tiles ct = blockIdx.y*4 + nt.
// ---------------------------------------------------------------------------
__global__ __launch_bounds__(128, 1)
void xproj_wmma(const bf16_t* __restrict__ Xhi, const bf16_t* __restrict__ Xlo,
                const bf16_t* __restrict__ Whi, const bf16_t* __restrict__ Wlo,
                const float* __restrict__ bih, const float* __restrict__ bhh,
                float* __restrict__ Pf) {
    const int tid    = threadIdx.x;
    const int lane   = tid & 31;
    const int laneM  = lane & 15;
    const int laneHi = lane >> 4;
    const int wave   = tid >> 5;

    const int t  = blockIdx.x;
    const int r  = wave;
    const int c0 = blockIdx.y * 4;

    v8f accHH[4] = {}, accHL[4] = {}, accLH[4] = {};
    const long arowoff = ((long)t * 64 + r * 16 + laneM) * ISZ;
    const bf16_t* ah = Xhi + arowoff;
    const bf16_t* al = Xlo + arowoff;
    const bf16_t* brecH[4];
    const bf16_t* brecL[4];
#pragma unroll
    for (int nt = 0; nt < 4; ++nt) {
        const long off = (long)(c0 + nt) * (NKT * 512) + lane * 16;
        brecH[nt] = Whi + off;
        brecL[nt] = Wlo + off;
    }

#pragma unroll 2
    for (int kt = 0; kt < NKT; ++kt) {
        const v16bf ahi = load_afrag(ah, kt * 32, laneHi);
        const v16bf alo = load_afrag(al, kt * 32, laneHi);
#pragma unroll
        for (int nt = 0; nt < 4; ++nt) {
            const v16bf bhi = load_bfrag(brecH[nt] + kt * 512);
            const v16bf blo = load_bfrag(brecL[nt] + kt * 512);
            accHH[nt] = wmma_bf16(ahi, bhi, accHH[nt]);
            accHL[nt] = wmma_bf16(ahi, blo, accHL[nt]);
            accLH[nt] = wmma_bf16(alo, bhi, accLH[nt]);
        }
    }

    // store in f32 frag layout (2 x b128 per tile), bias folded in
#pragma unroll
    for (int nt = 0; nt < 4; ++nt) {
        const int ct   = c0 + nt;
        const float bs = bih[ct * 16 + laneM] + bhh[ct * 16 + laneM];
        float* dst = Pf + (((long)t * 4 + r) * NCT + ct) * 256 + lane * 8;
        v4f lo4, hi4;
#pragma unroll
        for (int v = 0; v < 4; ++v)
            lo4[v] = accHH[nt][v] + (accHL[nt][v] + accLH[nt][v]) + bs;
#pragma unroll
        for (int v = 0; v < 4; ++v)
            hi4[v] = accHH[nt][v + 4] + (accHL[nt][v + 4] + accLH[nt][v + 4]) + bs;
        *(v4f*)(dst)     = lo4;
        *(v4f*)(dst + 4) = hi4;
    }
}

// ---------------------------------------------------------------------------
// Phase 2: persistent scan. 32 blocks x 128 threads = 128 waves.
// wave w: r = w>>5 (batch tile), j = w&31 (h-col tile); gate g col tile is
// g*32 + j. h kept as bf16 hi/lo; 12 independent WMMA chains of depth 16.
// ---------------------------------------------------------------------------
__global__ __launch_bounds__(128, 1)
void lstm_scan_wmma(const float* __restrict__ Pf,
                    const bf16_t* __restrict__ Whi, const bf16_t* __restrict__ Wlo,
                    bf16_t* __restrict__ hAhi, bf16_t* __restrict__ hAlo,
                    bf16_t* __restrict__ hBhi, bf16_t* __restrict__ hBlo,
                    float* __restrict__ c_buf, float* __restrict__ out,
                    unsigned* __restrict__ cnt) {
    const int tid    = threadIdx.x;
    const int lane   = tid & 31;
    const int laneM  = lane & 15;
    const int laneHi = lane >> 4;
    const int w      = blockIdx.x * 4 + (tid >> 5);  // 0..127
    const int r      = w >> 5;                       // batch tile 0..3
    const int j      = w & 31;                       // h-col tile 0..31
    const int R      = r * 16;
    const int colH   = j * 16 + laneM;

    const bf16_t* brecH[4];
    const bf16_t* brecL[4];
#pragma unroll
    for (int g = 0; g < 4; ++g) {
        const long off = (long)(g * 32 + j) * (NKT * 512) + lane * 16;
        brecH[g] = Whi + off;
        brecL[g] = Wlo + off;
    }

    for (int t = 0; t < T_STEPS; ++t) {
        const bf16_t* phi = (t & 1) ? hBhi : hAhi;
        const bf16_t* plo = (t & 1) ? hBlo : hAlo;
        bf16_t* nhi = (t & 1) ? hAhi : hBhi;
        bf16_t* nlo = (t & 1) ? hAlo : hBlo;

        v8f accHH[4] = {}, accHL[4] = {}, accLH[4] = {};
        const long arowoff = (long)(R + laneM) * HSZ;
        const bf16_t* ah = phi + arowoff;
        const bf16_t* al = plo + arowoff;

#pragma unroll 2
        for (int kt = 0; kt < NKT; ++kt) {
            const v16bf ahi = load_afrag(ah, kt * 32, laneHi);
            const v16bf alo = load_afrag(al, kt * 32, laneHi);
#pragma unroll
            for (int g = 0; g < 4; ++g) {
                const v16bf bhi = load_bfrag(brecH[g] + kt * 512);
                const v16bf blo = load_bfrag(brecL[g] + kt * 512);
                accHH[g] = wmma_bf16(ahi, bhi, accHH[g]);  // 12 indep. chains
                accHL[g] = wmma_bf16(ahi, blo, accHL[g]);
                accLH[g] = wmma_bf16(alo, bhi, accLH[g]);
            }
        }

        // gate pre-activations: acc + x_proj (frag-layout b128 loads)
        v8f gate[4];
#pragma unroll
        for (int g = 0; g < 4; ++g) {
            const float* rec =
                Pf + (((long)t * 4 + r) * NCT + (g * 32 + j)) * 256 + lane * 8;
            v4f plo4 = *(const v4f*)(rec);
            v4f phi4 = *(const v4f*)(rec + 4);
#pragma unroll
            for (int v = 0; v < 4; ++v)
                gate[g][v] = accHH[g][v] + (accHL[g][v] + accLH[g][v]) + plo4[v];
#pragma unroll
            for (int v = 0; v < 4; ++v)
                gate[g][v + 4] =
                    accHH[g][v + 4] + (accHL[g][v + 4] + accLH[g][v + 4]) + phi4[v];
        }

        // LSTM cell update; producer-side bf16 split of h_new
#pragma unroll
        for (int v = 0; v < 8; ++v) {
            const int brow = R + v + 8 * laneHi;
            const float ig = 1.0f / (1.0f + __expf(-gate[0][v]));
            const float fg = 1.0f / (1.0f + __expf(-gate[1][v]));
            const float g_ = tanhf(gate[2][v]);
            const float og = 1.0f / (1.0f + __expf(-gate[3][v]));

            const long cidx = (long)brow * HSZ + colH;
            const float cn  = fg * c_buf[cidx] + ig * g_;
            const float hnv = og * tanhf(cn);
            c_buf[cidx] = cn;

            const bf16_t hh = (bf16_t)hnv;
            nhi[cidx] = hh;
            nlo[cidx] = (bf16_t)(hnv - (float)hh);

            if (t == T_STEPS - 1) {
                out[cidx]               = hnv;  // hy
                out[BATCH * HSZ + cidx] = cn;   // cy
            }
        }

        // device-wide barrier between timesteps (monotonic counter)
        __syncthreads();
        if (tid == 0) {
            __threadfence();
            __hip_atomic_fetch_add(cnt, 1u, __ATOMIC_ACQ_REL,
                                   __HIP_MEMORY_SCOPE_AGENT);
            const unsigned target = (unsigned)(t + 1) * 32u;
            while (__hip_atomic_load(cnt, __ATOMIC_ACQUIRE,
                                     __HIP_MEMORY_SCOPE_AGENT) < target) {
                __builtin_amdgcn_s_sleep(1);
            }
        }
        __syncthreads();
    }
}

// ---------------------------------------------------------------------------
// Host launcher.
// d_in: input, hx, cx, wih, whh, bih, bhh (f32). d_out: hy | cy (f32).
// ws: cnt(256B) | c(128KB) | 4x h bf16(64KB) | 4x Wpack bf16(2MB)
//     | Xhi,Xlo bf16(32MB each) | Pf f32(256MB)
// ---------------------------------------------------------------------------
extern "C" void kernel_launch(void* const* d_in, const int* in_sizes, int n_in,
                              void* d_out, int out_size, void* d_ws, size_t ws_size,
                              hipStream_t stream) {
    const float* X   = (const float*)d_in[0];
    const float* hx  = (const float*)d_in[1];
    const float* cx  = (const float*)d_in[2];
    const float* wih = (const float*)d_in[3];
    const float* whh = (const float*)d_in[4];
    const float* bih = (const float*)d_in[5];
    const float* bhh = (const float*)d_in[6];
    float* out = (float*)d_out;

    char* p = (char*)d_ws;
    unsigned* cnt = (unsigned*)p;                 p += 256;
    float* c_b    = (float*)p;                    p += (size_t)BATCH * HSZ * 4;
    bf16_t* hAhi  = (bf16_t*)p;                   p += (size_t)BATCH * HSZ * 2;
    bf16_t* hAlo  = (bf16_t*)p;                   p += (size_t)BATCH * HSZ * 2;
    bf16_t* hBhi  = (bf16_t*)p;                   p += (size_t)BATCH * HSZ * 2;
    bf16_t* hBlo  = (bf16_t*)p;                   p += (size_t)BATCH * HSZ * 2;
    bf16_t* W1hi  = (bf16_t*)p;                   p += (size_t)WPACK_ELEMS * 2;
    bf16_t* W1lo  = (bf16_t*)p;                   p += (size_t)WPACK_ELEMS * 2;
    bf16_t* W2hi  = (bf16_t*)p;                   p += (size_t)WPACK_ELEMS * 2;
    bf16_t* W2lo  = (bf16_t*)p;                   p += (size_t)WPACK_ELEMS * 2;
    const int NX  = T_STEPS * BATCH * ISZ;        // 16,777,216
    bf16_t* Xhi   = (bf16_t*)p;                   p += (size_t)NX * 2;
    bf16_t* Xlo   = (bf16_t*)p;                   p += (size_t)NX * 2;
    float* Pf     = (float*)p;                    // 256 MiB

    hipMemsetAsync(d_ws, 0, 256, stream);
    hipMemcpyAsync(c_b, cx, (size_t)BATCH * HSZ * sizeof(float),
                   hipMemcpyDeviceToDevice, stream);

    split_f32_bf16<<<NX / 256, 256, 0, stream>>>(X, Xhi, Xlo, NX);
    split_f32_bf16<<<(BATCH * HSZ) / 256, 256, 0, stream>>>(hx, hAhi, hAlo,
                                                            BATCH * HSZ);
    pack_w_bf16<<<WPACK_ELEMS / 256, 256, 0, stream>>>(wih, W1hi, W1lo);
    pack_w_bf16<<<WPACK_ELEMS / 256, 256, 0, stream>>>(whh, W2hi, W2lo);

    dim3 g1(T_STEPS, 32);
    xproj_wmma<<<g1, 128, 0, stream>>>(Xhi, Xlo, W1hi, W1lo, bih, bhh, Pf);

    lstm_scan_wmma<<<32, 128, 0, stream>>>(Pf, W2hi, W2lo, hAhi, hAlo, hBhi,
                                           hBlo, c_b, out, cnt);
}
